// HeteroGAT_8770323218779
// MI455X (gfx1250) — matrix operand order repaired and checked by reading, not compiled, not run on hardware
//
#include <hip/hip_runtime.h>
#include <cstdint>

#define NN 30000
#define EE 300000
#define HD 128
#define HH 4
#define DD 32
#define CC 16
#define TT 2
#define SLOPE 0.2f
#define MT_PER_BLOCK 25   // 1875 row tiles = 75 blocks * 25

typedef __attribute__((ext_vector_type(16))) __bf16 v16bf;
typedef __attribute__((ext_vector_type(8)))  float v8f;
typedef __attribute__((ext_vector_type(4)))  unsigned int uivec4;
typedef __attribute__((ext_vector_type(4)))  unsigned int v4ui;
typedef __attribute__((ext_vector_type(8)))  int v8i;
typedef __attribute__((ext_vector_type(4)))  int v4i;

union Frag { v16bf v; uivec4 q[2]; };

__device__ __forceinline__ unsigned short f2bf(float f) {
  unsigned int u = __float_as_uint(f);
  unsigned int r = 0x7FFFu + ((u >> 16) & 1u);   // round-to-nearest-even
  return (unsigned short)((u + r) >> 16);
}
// order-preserving float -> uint encoding for atomicMax
__device__ __forceinline__ unsigned int fenc(float f) {
  unsigned int u = __float_as_uint(f);
  return (u & 0x80000000u) ? ~u : (u | 0x80000000u);
}
__device__ __forceinline__ float fdec(unsigned int u) {
  return __uint_as_float((u & 0x80000000u) ? (u ^ 0x80000000u) : ~u);
}

__global__ void k_f2bf(const float* __restrict__ src, unsigned short* __restrict__ dst, int n) {
  int i = blockIdx.x * blockDim.x + threadIdx.x;
  if (i < n) dst[i] = f2bf(src[i]);
}

// Wt[n*K + k] = bf16(W[k*cols + n])  (transpose + convert; K rows, `cols` columns)
__global__ void k_wtr(const float* __restrict__ W, unsigned short* __restrict__ Wt, int K, int cols) {
  int i = blockIdx.x * blockDim.x + threadIdx.x;
  if (i >= K * cols) return;
  int n = i / K, k = i % K;
  Wt[i] = f2bf(W[k * cols + n]);
}

__global__ void k_zero(float* __restrict__ p, int n) {
  int i = blockIdx.x * blockDim.x + threadIdx.x;
  if (i < n) p[i] = 0.0f;
}

__global__ void k_init_ms(unsigned int* __restrict__ m, float* __restrict__ s, int n) {
  int i = blockIdx.x * blockDim.x + threadIdx.x;
  if (i < n) { m[i] = fenc(-__builtin_inff()); s[i] = 0.0f; }
}

// ---------------------------------------------------------------------------
// Wide GEMM: C[M,128] = A[M,128](bf16) @ Bt[128,128](bf16, = W^T).
// 8 waves/block: stage W^T (32KB) in LDS once via the Tensor Data Mover,
// each wave keeps its column-tile's 4 B-fragments in registers across 25 row
// tiles; A streamed; C stored through one base pointer + const offsets.
// grid = 75, block = 256.
// ---------------------------------------------------------------------------
__global__ void k_gemm_wide(const unsigned short* __restrict__ A,
                            const unsigned short* __restrict__ Bt,
                            float* __restrict__ Cmat) {
  __shared__ unsigned short lds_w[HD * HD];   // 32 KB

#if __has_builtin(__builtin_amdgcn_tensor_load_to_lds)
  // Tensor Data Mover: one descriptor moves the whole 128x128x2B tile.
  if (threadIdx.x < 32) {                     // wave 0 issues the DMA
    unsigned long long ga = (unsigned long long)(uintptr_t)Bt;
    unsigned int lds_off = (unsigned int)(uintptr_t)(void*)lds_w;  // addr[31:0] = LDS offset
    v4ui g0;
    g0[0] = 1u;                                   // count=1, is_restore=0, gather off
    g0[1] = lds_off;                              // lds_addr
    g0[2] = (unsigned int)(ga & 0xFFFFFFFFu);     // global_addr[31:0]
    g0[3] = (unsigned int)((ga >> 32) & 0x01FFFFFFu) | (2u << 30);  // addr[56:32] | type=2
    v8i g1;
    g1[0] = 0x00010000;          // wg_mask=0, data_size=1 (2 bytes)
    g1[1] = (int)(128u << 16);   // tensor_dim0[15:0]=128 (bits[63:48])
    g1[2] = (int)(128u << 16);   // tensor_dim1[15:0]=128 (bits[95:80])
    g1[3] = (int)(128u << 16);   // tile_dim0=128 (bits[127:112])
    g1[4] = 128;                 // tile_dim1=128 (bits[143:128])
    g1[5] = 128;                 // tensor_dim0_stride[31:0]=128
    g1[6] = 0;
    g1[7] = 0;
    v4i g2 = {};                 // 2D tensor: groups 2/3 unused
    v4i g3 = {};
    v8i g4 = {};                 // extra group (6-arg toolchain), unused
    __builtin_amdgcn_tensor_load_to_lds(g0, g1, g2, g3, g4, 0);
    __builtin_amdgcn_s_wait_tensorcnt(0);
  }
  __syncthreads();
#else
  for (int i = threadIdx.x; i < (HD * HD) / 8; i += 256)
    ((uivec4*)lds_w)[i] = ((const uivec4*)Bt)[i];
  __syncthreads();
#endif

  const int wave = threadIdx.x >> 5;   // column tile 0..7
  const int lane = threadIdx.x & 31;
  const int l15  = lane & 15;
  const int hi   = (lane >> 4) & 1;

  // B fragments (held in registers for all 25 row tiles): ds_load_b128 x8
  Frag b[4];
  const unsigned short* bcol = lds_w + (size_t)(wave * 16 + l15) * 128;
#pragma unroll
  for (int kk = 0; kk < 4; ++kk) {
    const int k0 = kk * 32;
    b[kk].q[0] = *(const uivec4*)(bcol + k0 + hi * 8);
    b[kk].q[1] = *(const uivec4*)(bcol + k0 + 16 + hi * 8);
  }

  const int rbase = blockIdx.x * MT_PER_BLOCK * 16;
  const unsigned short* arow = A + (size_t)(rbase + l15) * 128;
  float* crow = Cmat + (size_t)(rbase + hi * 8) * HD + wave * 16 + l15;

  for (int mt = 0; mt < MT_PER_BLOCK; ++mt) {
    v8f acc = {};
#pragma unroll
    for (int kk = 0; kk < 4; ++kk) {
      const int k0 = kk * 32;
      Frag a;
      // 16-bit A 16x32 layout: lanes<16 hold K {0..7,16..23}, lanes>=16 hold K {8..15,24..31}
      a.q[0] = *(const uivec4*)(arow + k0 + hi * 8);
      a.q[1] = *(const uivec4*)(arow + k0 + 16 + hi * 8);
      acc = __builtin_amdgcn_wmma_f32_16x16x32_bf16(false, a.v, false, b[kk].v,
                                                    (short)0, acc, false, false);
    }
    // C/D layout: VGPR r -> row rbase+r (lanes<16) / rbase+8+r (lanes>=16);
    // consecutive rows are +512B -> constant instruction offsets off one base.
#pragma unroll
    for (int r = 0; r < 8; ++r) crow[(size_t)r * HD] = acc[r];
    arow += 16 * 128;
    crow += 16 * HD;
  }
}

// Narrow GEMM (final linear 128->16, fused bias): one wave per 16x16 tile.
__global__ void k_gemm16(const unsigned short* __restrict__ A,
                         const unsigned short* __restrict__ Bt,
                         float* __restrict__ Cmat,
                         const float* __restrict__ bias) {
  const int lane = threadIdx.x;
  const int row0 = blockIdx.x * 16;
  const int l15  = lane & 15;
  const int hi   = (lane >> 4) & 1;

  v8f acc = {};
  const unsigned short* arow = A  + (size_t)(row0 + l15) * 128;
  const unsigned short* bcol = Bt + (size_t)l15 * 128;
#pragma unroll
  for (int k0 = 0; k0 < 128; k0 += 32) {
    Frag a, b;
    a.q[0] = *(const uivec4*)(arow + k0 + hi * 8);
    a.q[1] = *(const uivec4*)(arow + k0 + 16 + hi * 8);
    b.q[0] = *(const uivec4*)(bcol + k0 + hi * 8);
    b.q[1] = *(const uivec4*)(bcol + k0 + 16 + hi * 8);
    acc = __builtin_amdgcn_wmma_f32_16x16x32_bf16(false, a.v, false, b.v,
                                                  (short)0, acc, false, false);
  }
  const float bv = bias[l15];
  float* crow = Cmat + (size_t)(row0 + hi * 8) * CC + l15;
#pragma unroll
  for (int r = 0; r < 8; ++r) crow[(size_t)r * CC] = acc[r] + bv;
}

// el/er[n,h] = sum_d feat[n,h,d] * al/ar[h,d]
__global__ void k_coeff(const float* __restrict__ feat,
                        const float* __restrict__ al, const float* __restrict__ ar,
                        float* __restrict__ el, float* __restrict__ er) {
  int i = blockIdx.x * blockDim.x + threadIdx.x;
  if (i >= NN * HH) return;
  int n = i >> 2, h = i & 3;
  const float* f = feat + (size_t)n * HD + h * DD;
  const float* a = al + h * DD;
  const float* b = ar + h * DD;
  float sl = 0.f, sr = 0.f;
#pragma unroll
  for (int d = 0; d < DD; ++d) { float fv = f[d]; sl += fv * a[d]; sr += fv * b[d]; }
  el[i] = sl; er[i] = sr;
}

__global__ void k_edge_logits(const float* __restrict__ el, const float* __restrict__ er,
                              const int* __restrict__ src, const int* __restrict__ dst,
                              float* __restrict__ ebuf, unsigned int* __restrict__ mmax) {
  int i = blockIdx.x * blockDim.x + threadIdx.x;
  if (i >= EE * HH) return;
  int e = i >> 2, h = i & 3;
  int sn = src[e], dn = dst[e];
  float v = el[sn * HH + h] + er[dn * HH + h];
  v = (v > 0.f) ? v : SLOPE * v;                 // leaky relu
  ebuf[i] = v;
  atomicMax(&mmax[dn * HH + h], fenc(v));        // segment max
}

__global__ void k_edge_exp(float* __restrict__ ebuf, const int* __restrict__ dst,
                           const unsigned int* __restrict__ mmax, float* __restrict__ ssum) {
  int i = blockIdx.x * blockDim.x + threadIdx.x;
  if (i >= EE * HH) return;
  int e = i >> 2, h = i & 3;
  int dn = dst[e];
  float ex = __expf(ebuf[i] - fdec(mmax[dn * HH + h]));
  ebuf[i] = ex;
  atomicAdd(&ssum[dn * HH + h], ex);             // segment sum
}

// out_acc[dst,h,d] += feat[src,h,d] * (ex / s[dst,h]);  thread per (edge, d), loop heads
__global__ void k_edge_scatter(const float* __restrict__ feat, const float* __restrict__ ebuf,
                               const float* __restrict__ ssum,
                               const int* __restrict__ src, const int* __restrict__ dst,
                               float* __restrict__ outacc) {
  int i = blockIdx.x * blockDim.x + threadIdx.x;
  if (i >= EE * DD) return;
  int e = i >> 5, d = i & 31;
  int sn = src[e], dn = dst[e];
  __builtin_prefetch(feat + (size_t)sn * HD, 0, 0);   // global_prefetch_b8 on gathered row
#pragma unroll
  for (int h = 0; h < HH; ++h) {
    float alpha = ebuf[e * HH + h] / ssum[dn * HH + h];
    atomicAdd(&outacc[(size_t)dn * HD + h * DD + d],
              feat[(size_t)sn * HD + h * DD + d] * alpha);
  }
}

// hb[n,c] = bf16( maybe_relu(acc[n,c] + b_t0[c] + b_t1[c]) )
__global__ void k_finalize(const float* __restrict__ acc, const float* __restrict__ b0,
                           const float* __restrict__ b1, unsigned short* __restrict__ hb,
                           int relu) {
  int i = blockIdx.x * blockDim.x + threadIdx.x;
  if (i >= NN * HD) return;
  int c = i & (HD - 1);
  float v = acc[i] + b0[c] + b1[c];
  if (relu) v = fmaxf(v, 0.f);
  hb[i] = f2bf(v);
}

extern "C" void kernel_launch(void* const* d_in, const int* in_sizes, int n_in,
                              void* d_out, int out_size, void* d_ws, size_t ws_size,
                              hipStream_t stream) {
  (void)in_sizes; (void)n_in; (void)out_size; (void)ws_size;
  const float* x     = (const float*)d_in[0];
  const int*   src[2] = {(const int*)d_in[1], (const int*)d_in[3]};
  const int*   dst[2] = {(const int*)d_in[2], (const int*)d_in[4]};
  const float* W[3]  = {(const float*)d_in[5],  (const float*)d_in[9],  (const float*)d_in[13]};
  const float* al[3] = {(const float*)d_in[6],  (const float*)d_in[10], (const float*)d_in[14]};
  const float* ar[3] = {(const float*)d_in[7],  (const float*)d_in[11], (const float*)d_in[15]};
  const float* bb[3] = {(const float*)d_in[8],  (const float*)d_in[12], (const float*)d_in[16]};
  const float* Wout  = (const float*)d_in[17];
  const float* bout  = (const float*)d_in[18];
  float* out = (float*)d_out;

  char* ws = (char*)d_ws;
  size_t off = 0;
  auto salloc = [&](size_t bytes) {
    void* p = ws + off; off = (off + bytes + 255) & ~(size_t)255; return p;
  };
  unsigned short* hb  = (unsigned short*)salloc((size_t)NN * HD * 2);   // bf16 features
  float* feat         = (float*)salloc((size_t)NN * HD * 4);            // per-etype h@W
  float* outacc       = (float*)salloc((size_t)NN * HD * 4);            // layer accumulator
  float* el           = (float*)salloc((size_t)NN * HH * 4);
  float* er           = (float*)salloc((size_t)NN * HH * 4);
  unsigned int* mmax  = (unsigned int*)salloc((size_t)NN * HH * 4);
  float* ssum         = (float*)salloc((size_t)NN * HH * 4);
  float* ebuf         = (float*)salloc((size_t)EE * HH * 4);
  unsigned short* Wt  = (unsigned short*)salloc((size_t)3 * TT * HD * HD * 2); // W^T bf16
  unsigned short* WoT = (unsigned short*)salloc((size_t)CC * HD * 2);          // Wout^T bf16

  auto cdiv = [](int a, int b) { return (a + b - 1) / b; };

  // one-time conversions (tiny)
  k_f2bf<<<cdiv(NN * HD, 256), 256, 0, stream>>>(x, hb, NN * HD);
  for (int l = 0; l < 3; ++l)
    for (int t = 0; t < TT; ++t)
      k_wtr<<<cdiv(HD * HD, 256), 256, 0, stream>>>(
          W[l] + (size_t)t * HD * HD, Wt + (size_t)(l * TT + t) * HD * HD, HD, HD);
  k_wtr<<<cdiv(HD * CC, 256), 256, 0, stream>>>(Wout, WoT, HD, CC);

  for (int l = 0; l < 3; ++l) {
    k_zero<<<cdiv(NN * HD, 256), 256, 0, stream>>>(outacc, NN * HD);
    for (int t = 0; t < TT; ++t) {
      k_gemm_wide<<<(NN / 16) / MT_PER_BLOCK, 256, 0, stream>>>(
          hb, Wt + (size_t)(l * TT + t) * HD * HD, feat);
      k_coeff<<<cdiv(NN * HH, 256), 256, 0, stream>>>(
          feat, al[l] + t * HH * DD, ar[l] + t * HH * DD, el, er);
      k_init_ms<<<cdiv(NN * HH, 256), 256, 0, stream>>>(mmax, ssum, NN * HH);
      k_edge_logits<<<cdiv(EE * HH, 256), 256, 0, stream>>>(el, er, src[t], dst[t], ebuf, mmax);
      k_edge_exp<<<cdiv(EE * HH, 256), 256, 0, stream>>>(ebuf, dst[t], mmax, ssum);
      k_edge_scatter<<<cdiv(EE * DD, 256), 256, 0, stream>>>(
          feat, ebuf, ssum, src[t], dst[t], outacc);
    }
    k_finalize<<<cdiv(NN * HD, 256), 256, 0, stream>>>(
        outacc, bb[l], bb[l] + HD, hb, (l < 2) ? 1 : 0);
  }
  // final linear: [N,128] @ [128,16] + bout  -> d_out (f32)
  k_gemm16<<<NN / 16, 32, 0, stream>>>(hb, WoT, out, bout);
}
